// SelfAttention_29626684408350
// MI455X (gfx1250) — matrix operand order repaired and checked by reading, not compiled, not run on hardware
//
#include <hip/hip_runtime.h>
#include <hip/hip_bf16.h>
#include <math.h>

typedef __attribute__((ext_vector_type(16))) __bf16 v16bf;
typedef __attribute__((ext_vector_type(8)))  __bf16 v8bf;
typedef __attribute__((ext_vector_type(8)))  float  v8f;

static constexpr int Bn = 8, Sn = 2048, Dn = 512, Mn = Bn * Sn;

__device__ __forceinline__ v8f wmma_bf16(v16bf a, v16bf b, v8f c) {
  // (neg_a, A, neg_b, B, c_mod, C, reuse_a, reuse_b)
  return __builtin_amdgcn_wmma_f32_16x16x32_bf16(false, a, false, b, (short)0, c,
                                                 false, false);
}

// A-matrix fragment (16x32 bf16, MxK). Lane L: m = L&15, h = L>>4.
// Lane holds 16 consecutive K values: K = 16h..16h+15 (matches load_b_frag).
__device__ __forceinline__ v16bf load_a_frag(const __bf16* rowPtr, int h) {
  v8bf c0 = *(const v8bf*)(rowPtr + h * 16);
  v8bf c1 = *(const v8bf*)(rowPtr + h * 16 + 8);
  v16bf r;
#pragma unroll
  for (int i = 0; i < 8; ++i) { r[i] = c0[i]; r[8 + i] = c1[i]; }
  return r;
}

// B-matrix fragment (32x16 bf16, KxN). Lane L: n = L&15, kh = L>>4.
// Lane holds 16 consecutive K values of column n; p points at B^T[n][k0 + kh*16].
__device__ __forceinline__ v16bf load_b_frag(const __bf16* p) {
  v8bf c0 = *(const v8bf*)(p);
  v8bf c1 = *(const v8bf*)(p + 8);
  v16bf r;
#pragma unroll
  for (int i = 0; i < 8; ++i) { r[i] = c0[i]; r[8 + i] = c1[i]; }
  return r;
}

// ---- CDNA5 async Global->LDS copy (ASYNCcnt path), inline asm ---------------
// lds_off: byte offset into LDS (low 32 bits of a generic shared pointer).
// gaddr  : 64-bit global address. Copies 16 bytes per lane, asynchronously.
__device__ __forceinline__ void async_g2l_b128(unsigned lds_off, const void* gp) {
  unsigned long long ga = (unsigned long long)(uintptr_t)gp;
  asm volatile("global_load_async_to_lds_b128 %0, %1, off"
               :
               : "v"(lds_off), "v"(ga)
               : "memory");
}
#define WAIT_ASYNCCNT(N) asm volatile("s_wait_asynccnt " #N ::: "memory")

// ---------------------------------------------------------------- fp32 -> bf16
__global__ void cvt_kernel(const float* __restrict__ in, __bf16* __restrict__ out,
                           int n) {
  int i = blockIdx.x * blockDim.x + threadIdx.x;
  if (i < n) out[i] = (__bf16)in[i];
}

// -------------------------------------------------- Y = (X @ W^T + b) * scale
__global__ void proj_gemm_kernel(const __bf16* __restrict__ X,
                                 const __bf16* __restrict__ W,
                                 const float* __restrict__ bias,
                                 __bf16* __restrict__ Y, float scale,
                                 int transposeOut) {
  const int tid = threadIdx.x;
  const int w = tid >> 5, lane = tid & 31;
  const int wm = w & 3, wn = w >> 2;
  const int m_base = blockIdx.x * 128 + wm * 32;
  const int n_base = blockIdx.y * 64 + wn * 32;
  const int m = lane & 15, h = lane >> 4;

  v8f acc[2][2] = {};
  for (int dc = 0; dc < Dn / 32; ++dc) {
    const int d0 = dc * 32;
    v16bf a0 = load_a_frag(X + (size_t)(m_base + m) * Dn + d0, h);
    v16bf a1 = load_a_frag(X + (size_t)(m_base + 16 + m) * Dn + d0, h);
    v16bf b0 = load_b_frag(W + (size_t)(n_base + m) * Dn + d0 + h * 16);
    v16bf b1 = load_b_frag(W + (size_t)(n_base + 16 + m) * Dn + d0 + h * 16);
    acc[0][0] = wmma_bf16(a0, b0, acc[0][0]);
    acc[0][1] = wmma_bf16(a0, b1, acc[0][1]);
    acc[1][0] = wmma_bf16(a1, b0, acc[1][0]);
    acc[1][1] = wmma_bf16(a1, b1, acc[1][1]);
  }
#pragma unroll
  for (int mi = 0; mi < 2; ++mi)
#pragma unroll
    for (int ni = 0; ni < 2; ++ni) {
      const int n_out = n_base + ni * 16 + m;
      const float bv = bias[n_out];
#pragma unroll
      for (int r = 0; r < 8; ++r) {
        const int m_out = m_base + mi * 16 + r + 8 * h;  // C layout: M = r + 8h
        const float val = (acc[mi][ni][r] + bv) * scale;
        if (!transposeOut) {
          Y[(size_t)m_out * Dn + n_out] = (__bf16)val;
        } else {
          const int bb = m_out >> 11, ss = m_out & (Sn - 1);
          Y[((size_t)bb * Dn + n_out) * Sn + ss] = (__bf16)val;
        }
      }
    }
}

// ------------------------------------------------------- one-pass flash attn
// Grid: (Sn/128, Bn); 256 threads (8 waves). Q pre-scaled by 1/sqrt(D).
// K tiles (32 x 512 bf16) are double-buffered in LDS via async global->LDS DMA
// issued one tile ahead; score B-frags then come from ds_load_b128.
__global__ void flash_attn_kernel(const __bf16* __restrict__ Q,
                                  const __bf16* __restrict__ K,
                                  const __bf16* __restrict__ VT,
                                  float* __restrict__ out) {
  constexpr int QT = 128;   // q rows / block
  constexpr int KT = 32;    // k cols / iteration
  constexpr int NT = Sn / KT;
  constexpr int PST = 48;   // P LDS row stride (bf16): 96B, 16B-aligned chunks
  constexpr int KST = 528;  // K LDS row stride (bf16): 1056B, 16B-aligned
  __shared__ __attribute__((aligned(16))) __bf16 Kt[2][KT * KST];
  __shared__ __attribute__((aligned(16))) __bf16 Plds[QT * PST];
  __shared__ float alphaLds[QT];
  __shared__ float lLds[QT];

  const int tid = threadIdx.x;
  const int w = tid >> 5, lane = tid & 31;
  const int m = lane & 15, h = lane >> 4;
  const int b = blockIdx.y;
  const int q0 = blockIdx.x * QT;
  const int qw = q0 + w * 16;  // score-stage rows
  const int e0 = w * 64;       // PV-stage output columns

  const __bf16* Kbase = K + (size_t)b * Sn * Dn;  // [2048][512]

  // Issue one K tile per call: 32 rows x 1024B = 2048 x 16B chunks, 8/thread.
  auto issue_ktile = [&](int k0, int buf) {
    const unsigned ldsBase = (unsigned)(uintptr_t)&Kt[buf][0];
#pragma unroll
    for (int j = 0; j < 8; ++j) {
      const int c = tid + j * 256;          // chunk id 0..2047
      const int row = c >> 6, col = c & 63; // 64 x 16B chunks per row
      async_g2l_b128(ldsBase + (unsigned)(row * (KST * 2) + col * 16),
                     (const char*)(Kbase + (size_t)(k0 + row) * Dn) + col * 16);
    }
  };

  issue_ktile(0, 0);  // prefetch tile 0 (overlaps with Q fragment loads below)

  // Q fragments for this wave's 16 rows, all 16 d-chunks (128 VGPRs).
  v16bf qa[16];
#pragma unroll
  for (int dc = 0; dc < 16; ++dc)
    qa[dc] = load_a_frag(Q + ((size_t)b * Sn + qw + m) * Dn + dc * 32, h);

  v8f oac[8][4] = {};  // 128 q x 64 e accumulator (256 VGPRs)
  float m_run[8], l_run[8];
#pragma unroll
  for (int r = 0; r < 8; ++r) { m_run[r] = -INFINITY; l_run[r] = 0.f; }

  for (int kt = 0; kt < NT; ++kt) {
    const int k0 = kt * KT;
    const int cur = kt & 1;
    // Prefetch next tile into the other buffer, then wait for current tile
    // (async loads complete in order: allow the 8 newest to stay outstanding).
    if (kt + 1 < NT) {
      issue_ktile(k0 + KT, cur ^ 1);
      WAIT_ASYNCCNT(8);
    } else {
      WAIT_ASYNCCNT(0);
    }
    __syncthreads();

    // ---- scores: (16 q) x (32 k), contract over d; B-frags from LDS K tile.
    const __bf16* Kc = &Kt[cur][0];
    v8f st[2] = {};
#pragma unroll
    for (int dc = 0; dc < 16; ++dc) {
#pragma unroll
      for (int ks = 0; ks < 2; ++ks) {
        v16bf kb = load_b_frag(Kc + (ks * 16 + m) * KST + dc * 32 + h * 16);
        st[ks] = wmma_bf16(qa[dc], kb, st[ks]);
      }
    }
    // ---- online softmax over this wave's rows (row = r + 8h in C layout)
    float alpha[8];
#pragma unroll
    for (int r = 0; r < 8; ++r) {
      float v = fmaxf(st[0][r], st[1][r]);
#pragma unroll
      for (int off = 8; off >= 1; off >>= 1) v = fmaxf(v, __shfl_xor(v, off, 32));
      const float mnew = fmaxf(m_run[r], v);
      alpha[r] = __expf(m_run[r] - mnew);
      m_run[r] = mnew;
      const float p0 = __expf(st[0][r] - mnew);
      const float p1 = __expf(st[1][r] - mnew);
      st[0][r] = p0;
      st[1][r] = p1;
      float rs = p0 + p1;
#pragma unroll
      for (int off = 8; off >= 1; off >>= 1) rs += __shfl_xor(rs, off, 32);
      l_run[r] = l_run[r] * alpha[r] + rs;
    }
    // ---- publish P (bf16) and alpha
#pragma unroll
    for (int r = 0; r < 8; ++r) {
      const int row = w * 16 + r + 8 * h;
      Plds[row * PST + m] = (__bf16)st[0][r];
      Plds[row * PST + 16 + m] = (__bf16)st[1][r];
      if (m == 0) alphaLds[row] = alpha[r];
    }
    __syncthreads();
    // ---- rescale O by alpha, then O += P @ V (contract over k)
    v16bf vb[4];
#pragma unroll
    for (int et = 0; et < 4; ++et)
      vb[et] = load_b_frag(VT + ((size_t)b * Dn + e0 + et * 16 + m) * Sn + k0 + h * 16);
#pragma unroll
    for (int qs = 0; qs < 8; ++qs) {
#pragma unroll
      for (int r = 0; r < 8; ++r) {
        const float a = alphaLds[qs * 16 + r + 8 * h];
#pragma unroll
        for (int et = 0; et < 4; ++et) oac[qs][et][r] *= a;
      }
      v16bf pa = load_a_frag(&Plds[(qs * 16 + m) * PST], h);
#pragma unroll
      for (int et = 0; et < 4; ++et) oac[qs][et] = wmma_bf16(pa, vb[et], oac[qs][et]);
    }
    __syncthreads();  // protects Plds and the K buffer reused next iteration
  }
  // ---- share l, normalize, store fp32
#pragma unroll
  for (int r = 0; r < 8; ++r) {
    const int row = w * 16 + r + 8 * h;
    if (m == 0) lLds[row] = l_run[r];
  }
  __syncthreads();
#pragma unroll
  for (int qs = 0; qs < 8; ++qs) {
#pragma unroll
    for (int r = 0; r < 8; ++r) {
      const int row = qs * 16 + r + 8 * h;
      const float inv = 1.f / lLds[row];
#pragma unroll
      for (int et = 0; et < 4; ++et)
        out[((size_t)b * Sn + q0 + row) * Dn + e0 + et * 16 + m] = oac[qs][et][r] * inv;
    }
  }
}

extern "C" void kernel_launch(void* const* d_in, const int* in_sizes, int n_in,
                              void* d_out, int out_size, void* d_ws, size_t ws_size,
                              hipStream_t stream) {
  (void)in_sizes; (void)n_in; (void)out_size; (void)ws_size;
  const float* x  = (const float*)d_in[0];
  const float* Wq = (const float*)d_in[1];
  const float* bq = (const float*)d_in[2];
  const float* Wk = (const float*)d_in[3];
  const float* bk = (const float*)d_in[4];
  const float* Wv = (const float*)d_in[5];
  const float* bv = (const float*)d_in[6];
  float* out = (float*)d_out;

  char* ws = (char*)d_ws;
  size_t off = 0;
  auto alloc = [&](size_t bytes) -> void* {
    void* p = ws + off;
    off += (bytes + 255) & ~(size_t)255;
    return p;
  };
  __bf16* xb  = (__bf16*)alloc((size_t)Mn * Dn * 2);
  __bf16* wqb = (__bf16*)alloc((size_t)Dn * Dn * 2);
  __bf16* wkb = (__bf16*)alloc((size_t)Dn * Dn * 2);
  __bf16* wvb = (__bf16*)alloc((size_t)Dn * Dn * 2);
  __bf16* Qb  = (__bf16*)alloc((size_t)Mn * Dn * 2);
  __bf16* Kb  = (__bf16*)alloc((size_t)Mn * Dn * 2);
  __bf16* VTb = (__bf16*)alloc((size_t)Mn * Dn * 2);  // [B][D][S]

  const int nX = Mn * Dn, nW = Dn * Dn;
  cvt_kernel<<<(nX + 255) / 256, 256, 0, stream>>>(x, xb, nX);
  cvt_kernel<<<(nW + 255) / 256, 256, 0, stream>>>(Wq, wqb, nW);
  cvt_kernel<<<(nW + 255) / 256, 256, 0, stream>>>(Wk, wkb, nW);
  cvt_kernel<<<(nW + 255) / 256, 256, 0, stream>>>(Wv, wvb, nW);

  const dim3 pg(Mn / 128, Dn / 64);
  const float qscale = 0.04419417382415922f;  // 1/sqrt(512), folded into Q
  proj_gemm_kernel<<<pg, 256, 0, stream>>>(xb, wqb, bq, Qb, qscale, 0);
  proj_gemm_kernel<<<pg, 256, 0, stream>>>(xb, wkb, bk, Kb, 1.0f, 0);
  proj_gemm_kernel<<<pg, 256, 0, stream>>>(xb, wvb, bv, VTb, 1.0f, 1);

  flash_attn_kernel<<<dim3(Sn / 128, Bn), 256, 0, stream>>>(Qb, Kb, VTb, out);
}
